// LinTransformerModel_43224550868160
// MI455X (gfx1250) — compile-verified, matrix operand order checked by reading, not run on hardware
//
#include <hip/hip_runtime.h>
#include <hip/hip_bf16.h>
#include <math.h>

// ---------------- problem constants ----------------
#define SEQ_L 2048
#define BATCH_ 16
#define DIO 512
#define DM 1024
#define M_TOK (SEQ_L * BATCH_)   // 32768 tokens
#define NLAYER 3
#define NCHUNK 32
#define CLEN (SEQ_L / NCHUNK)    // 64 seq steps / chunk
#define CROWS (CLEN * BATCH_)    // 1024 rows / chunk

typedef __bf16 bf16_t;
typedef __attribute__((ext_vector_type(16))) __bf16 v16bf;
typedef __attribute__((ext_vector_type(8)))  __bf16 bf16x8;
typedef __attribute__((ext_vector_type(8)))  float  v8f;

union FragU { v16bf v; bf16x8 h[2]; bf16_t e[16]; };

// ISA 7.12.2: 16-bit A 16x32 — lane l holds row (l&15); lane-half H=l>>4 holds
// K in {H*8 + j (j<8)} and {H*8 + 8 + j (j>=8)} -> two contiguous 8-element runs.
__device__ inline int kmap_base(int lane) { return (lane >> 4) << 3; }

__device__ inline v16bf load_frag_rm(const bf16_t* base, int ld, int row0, int k0, int lane) {
  const bf16_t* p = base + (size_t)(row0 + (lane & 15)) * ld + k0 + kmap_base(lane);
  FragU f;
  f.h[0] = *(const bf16x8*)(p);
  f.h[1] = *(const bf16x8*)(p + 16);
  return f.v;
}
__device__ inline v16bf load_frag_cm(const bf16_t* base, int ld, int rc0, int k0, int lane) {
  FragU f;
  int rc = rc0 + (lane & 15);
  int kb = k0 + kmap_base(lane);
#pragma unroll
  for (int j = 0; j < 16; ++j) {
    int k = kb + (j < 8 ? j : j + 8);
    f.e[j] = base[(size_t)k * ld + rc];
  }
  return f.v;
}
__device__ inline v16bf load_frag_cm_f32(const float* base, int ld, int rc0, int k0, int lane) {
  FragU f;
  int rc = rc0 + (lane & 15);
  int kb = k0 + kmap_base(lane);
#pragma unroll
  for (int j = 0; j < 16; ++j) {
    int k = kb + (j < 8 ? j : j + 8);
    f.e[j] = (bf16_t)base[(size_t)k * ld + rc];
  }
  return f.v;
}

__device__ inline v8f wmma_bf16(v16bf a, v16bf b, v8f c) {
  return __builtin_amdgcn_wmma_f32_16x16x32_bf16(false, a, false, b, (short)0, c, false, false);
}

// CDNA5 async copy global->LDS (ASYNCcnt), 16B per lane (ISA 15.18.3 op 98).
__device__ inline void async_cp16(unsigned lds_off, const bf16_t* g) {
  asm volatile("global_load_async_to_lds_b128 %0, %1, off"
               :: "v"(lds_off), "v"((unsigned long long)(size_t)g) : "memory");
}
__device__ inline void wait_async0() {
  asm volatile("s_wait_asynccnt 0x0" ::: "memory");
}

// layouts / epilogues
enum { AL_RM = 0, AL_CM = 1 };
enum { BL_NN = 1, BL_NN_F32 = 2 };
enum {
  EPI_F32 = 0, EPI_F32_ACC, EPI_BF16, EPI_ELU1_BF16, EPI_BIAS_RELU_BF16,
  EPI_FFN2, EPI_IN_PROJ, EPI_MASK_BF16, EPI_BIAS_F32
};

template <int EPI>
__device__ inline void apply_epi(float val, int m, int n, size_t idx,
                                 float* outF, bf16_t* outB,
                                 const float* bias, const float* resid) {
  if constexpr (EPI == EPI_F32) {
    outF[idx] = val;
  } else if constexpr (EPI == EPI_F32_ACC) {
    outF[idx] += val;
  } else if constexpr (EPI == EPI_BF16) {
    outB[idx] = (bf16_t)val;
  } else if constexpr (EPI == EPI_ELU1_BF16) {
    val = val > 0.f ? val + 1.f : __expf(val);   // elu(x)+1
    outB[idx] = (bf16_t)val;
  } else if constexpr (EPI == EPI_BIAS_RELU_BF16) {
    val = fmaxf(val + bias[n], 0.f);
    outB[idx] = (bf16_t)val;
  } else if constexpr (EPI == EPI_FFN2) {
    val += bias[n] + resid[idx];
    outF[idx] = val;
    outB[idx] = (bf16_t)val;
  } else if constexpr (EPI == EPI_IN_PROJ) {
    val = fmaxf(val + bias[n], 0.f);
    const int s = m >> 4;  // token m = s*16+b
    const int i2 = n & ~1;
    const float div = __expf((float)i2 * -0.00899447302f);  // -ln(1e4)/1024
    const float ang = (float)s * div;
    val += (n & 1) ? __cosf(ang) : __sinf(ang);
    outF[idx] = val;
    outB[idx] = (bf16_t)val;
  } else if constexpr (EPI == EPI_MASK_BF16) {
    outB[idx] = ((n >> 4) <= (m >> 4)) ? (bf16_t)val : (bf16_t)0.f;
  } else if constexpr (EPI == EPI_BIAS_F32) {
    outF[idx] = val + bias[n];
  }
}

#define BLK_M 64
#define BLK_N 128
#define LPAD 40  // LDS row pitch (bf16 elems): 80B, 16B-aligned, 20-bank stride

// ---------------- LDS-staged NT GEMM (A row-major [M,K], B row-major [N,K]) --
// 64x128 block tile, 4 waves, each 32x64 (2x4 wmma accs). Double-buffered
// async global->LDS staging overlapped with WMMA; fragments via ds_load_b128.
template <int EPI>
__global__ __launch_bounds__(128) void gemm_nt_staged(
    const bf16_t* A, int lda, const bf16_t* B, int ldb,
    float* outF, bf16_t* outB, int ldd,
    const float* bias, const float* resid, int Kdim,
    long long aZStride, long long bZStride, long long oZStride) {
  __shared__ bf16_t sA[2][BLK_M * LPAD];
  __shared__ bf16_t sB[2][BLK_N * LPAD];
  const int tid = threadIdx.x;
  const int lane = tid & 31, wave = tid >> 5;
  const int mBlk = blockIdx.y * BLK_M, nBlk = blockIdx.x * BLK_N;
  const int wm = (wave >> 1) * 32, wn = (wave & 1) * 64;
  const int z = blockIdx.z;
  A += (long long)z * aZStride;
  B += (long long)z * bZStride;
  const long long oOff = (long long)z * oZStride;

  // stage one 32-wide k-slab: A 64x32 (256 16B segs), B 128x32 (512 16B segs)
  auto stage = [&](int buf, int k) {
#pragma unroll
    for (int it = 0; it < 2; ++it) {
      const int seg = tid + it * 128;
      const int r = seg >> 2, c = (seg & 3) * 8;
      async_cp16((unsigned)(size_t)&sA[buf][r * LPAD + c],
                 A + (size_t)(mBlk + r) * lda + k + c);
    }
#pragma unroll
    for (int it = 0; it < 4; ++it) {
      const int seg = tid + it * 128;
      const int r = seg >> 2, c = (seg & 3) * 8;
      async_cp16((unsigned)(size_t)&sB[buf][r * LPAD + c],
                 B + (size_t)(nBlk + r) * ldb + k + c);
    }
  };
  auto ldsFrag = [&](const bf16_t* s, int row0) -> v16bf {
    const bf16_t* p = s + (row0 + (lane & 15)) * LPAD + kmap_base(lane);
    FragU f;
    f.h[0] = *(const bf16x8*)(p);
    f.h[1] = *(const bf16x8*)(p + 16);
    return f.v;
  };

  v8f acc[2][4] = {};
  stage(0, 0);
  wait_async0();
  __syncthreads();
  int buf = 0;
  for (int k = 0; k < Kdim; k += 32) {
    if (k + 32 < Kdim) stage(buf ^ 1, k + 32);  // prefetch next slab (async)
    const v16bf a0 = ldsFrag(sA[buf], wm);
    const v16bf a1 = ldsFrag(sA[buf], wm + 16);
#pragma unroll
    for (int j = 0; j < 4; ++j) {
      const v16bf b = ldsFrag(sB[buf], wn + j * 16);
      acc[0][j] = wmma_bf16(a0, b, acc[0][j]);
      acc[1][j] = wmma_bf16(a1, b, acc[1][j]);
    }
    wait_async0();
    __syncthreads();
    buf ^= 1;
  }

#pragma unroll
  for (int i = 0; i < 2; ++i)
#pragma unroll
    for (int j = 0; j < 4; ++j) {
      const int ms = mBlk + wm + i * 16 + kmap_base(lane);
      const int n = nBlk + wn + j * 16 + (lane & 15);
#pragma unroll
      for (int v = 0; v < 8; ++v) {
        const int m = ms + v;
        const size_t idx = (size_t)oOff + (size_t)m * ldd + n;
        apply_epi<EPI>(acc[i][j][v], m, n, idx, outF, outB, bias, resid);
      }
    }
}

// ---------------- direct-load GEMM for transposed / K-major operands --------
template <int ALAY, int BLAY, int EPI>
__global__ __launch_bounds__(128) void gemm_wmma(
    const bf16_t* A, int lda, const void* Bv, int ldb,
    float* outF, bf16_t* outB, int ldd,
    const float* bias, const float* resid, int Kdim,
    long long aZStride, int aZOff, long long bZStride, int bZOff,
    long long oZStride, int oZOff) {
  const int lane = threadIdx.x & 31;
  const int wave = threadIdx.x >> 5;
  const int mBase = blockIdx.y * BLK_M + (wave >> 1) * 32;
  const int nBase = blockIdx.x * BLK_N + (wave & 1) * 64;
  const int z = blockIdx.z;

  A += (long long)(z + aZOff) * aZStride;
  const bf16_t* B = (const bf16_t*)Bv;
  const float* Bf = (const float*)Bv;
  const long long bOff = (long long)(z + bZOff) * bZStride;
  const long long oOff = (long long)(z + oZOff) * oZStride;

  v8f acc[2][4] = {};
  for (int k = 0; k < Kdim; k += 32) {
    v16bf a0, a1, b[4];
    if (ALAY == AL_RM) {
      a0 = load_frag_rm(A, lda, mBase, k, lane);
      a1 = load_frag_rm(A, lda, mBase + 16, k, lane);
    } else {
      a0 = load_frag_cm(A, lda, mBase, k, lane);
      a1 = load_frag_cm(A, lda, mBase + 16, k, lane);
    }
#pragma unroll
    for (int j = 0; j < 4; ++j) {
      if (BLAY == BL_NN)
        b[j] = load_frag_cm(B + bOff, ldb, nBase + j * 16, k, lane);
      else
        b[j] = load_frag_cm_f32(Bf + bOff, ldb, nBase + j * 16, k, lane);
    }
#pragma unroll
    for (int j = 0; j < 4; ++j) {
      acc[0][j] = wmma_bf16(a0, b[j], acc[0][j]);
      acc[1][j] = wmma_bf16(a1, b[j], acc[1][j]);
    }
  }

#pragma unroll
  for (int i = 0; i < 2; ++i)
#pragma unroll
    for (int j = 0; j < 4; ++j) {
      const int ms = mBase + i * 16 + kmap_base(lane);
      const int n = nBase + j * 16 + (lane & 15);
#pragma unroll
      for (int v = 0; v < 8; ++v) {
        const int m = ms + v;
        const size_t idx = (size_t)oOff + (size_t)m * ldd + n;
        apply_epi<EPI>(acc[i][j][v], m, n, idx, outF, outB, bias, resid);
      }
    }
}

// ---------------- small helper kernels ----------------
__global__ void cvt_f32_bf16(const float* src, bf16_t* dst, size_t n) {
  size_t i = (size_t)blockIdx.x * blockDim.x + threadIdx.x;
  if (i < n) dst[i] = (bf16_t)src[i];
}
__global__ void addip_f32(float* dst, const float* src, size_t n) {
  size_t i = (size_t)blockIdx.x * blockDim.x + threadIdx.x;
  if (i < n) dst[i] += src[i];
}
// denom[s,b] = q_{s,b} . cumsum_s(k)_{s,b}  — per-batch scan, 16 blocks
__global__ void denom_kernel(const bf16_t* Q, const bf16_t* K, float* denom) {
  const int b = blockIdx.x, tid = threadIdx.x;
  const int lane = tid & 31, wid = tid >> 5;
  __shared__ float red[8];
  float z[4] = {0.f, 0.f, 0.f, 0.f};
  for (int s = 0; s < SEQ_L; ++s) {
    const size_t row = ((size_t)s * BATCH_ + b) * DM + (size_t)tid * 4;
    float p = 0.f;
#pragma unroll
    for (int u = 0; u < 4; ++u) {
      z[u] += (float)K[row + u];
      p += (float)Q[row + u] * z[u];
    }
#pragma unroll
    for (int off = 16; off > 0; off >>= 1) p += __shfl_xor(p, off, 32);
    if (lane == 0) red[wid] = p;
    __syncthreads();
    if (tid == 0) {
      float t = 0.f;
#pragma unroll
      for (int w = 0; w < 8; ++w) t += red[w];
      denom[(size_t)s * BATCH_ + b] = t;
    }
    __syncthreads();
  }
}
__global__ void combine_kernel(const float* num, const float* denom, const float* h,
                               bf16_t* u, size_t n) {
  size_t i = (size_t)blockIdx.x * blockDim.x + threadIdx.x;
  if (i < n) u[i] = (bf16_t)(num[i] / denom[i >> 10] + h[i]);
}
__global__ void softmax_kernel(const float* logits, float* out) {
  const int t = blockIdx.x, tid = threadIdx.x;
  const int lane = tid & 31, wid = tid >> 5;
  __shared__ float red[8];
  const float* row = logits + (size_t)t * DIO;
  float a = row[tid], c = row[tid + 256];
  float m = fmaxf(a, c);
#pragma unroll
  for (int off = 16; off > 0; off >>= 1) m = fmaxf(m, __shfl_xor(m, off, 32));
  if (lane == 0) red[wid] = m;
  __syncthreads();
  float M = red[0];
#pragma unroll
  for (int w = 1; w < 8; ++w) M = fmaxf(M, red[w]);
  const float e0 = __expf(a - M), e1 = __expf(c - M);
  float s = e0 + e1;
#pragma unroll
  for (int off = 16; off > 0; off >>= 1) s += __shfl_xor(s, off, 32);
  __syncthreads();
  if (lane == 0) red[wid] = s;
  __syncthreads();
  float S = 0.f;
#pragma unroll
  for (int w = 0; w < 8; ++w) S += red[w];
  out[(size_t)t * DIO + tid] = e0 / S;
  out[(size_t)t * DIO + tid + 256] = e1 / S;
}

// ---------------- workspace layout (~705 MB) ----------------
static constexpr size_t SZ_WIN = (size_t)DM * DIO * 2;
static constexpr size_t SZ_WL = (size_t)NLAYER * DM * DM * 2;
static constexpr size_t SZ_WO = (size_t)DIO * DM * 2;
static constexpr size_t OFF_WIN = 0;
static constexpr size_t OFF_WQ = OFF_WIN + SZ_WIN;
static constexpr size_t OFF_WK = OFF_WQ + SZ_WL;
static constexpr size_t OFF_WV = OFF_WK + SZ_WL;
static constexpr size_t OFF_WH1 = OFF_WV + SZ_WL;
static constexpr size_t OFF_WH2 = OFF_WH1 + SZ_WL;
static constexpr size_t OFF_WO = OFF_WH2 + SZ_WL;
static constexpr size_t OFF_H = OFF_WO + SZ_WO;                       // f32 h
static constexpr size_t OFF_HB = OFF_H + (size_t)M_TOK * DM * 4;      // bf16 hb / scores / u
static constexpr size_t OFF_QB = OFF_HB + (size_t)M_TOK * DM * 2;     // bf16 Q / xb
static constexpr size_t OFF_KB = OFF_QB + (size_t)M_TOK * DM * 2;     // bf16 K / ffn-mid
static constexpr size_t OFF_VB = OFF_KB + (size_t)M_TOK * DM * 2;     // bf16 V
static constexpr size_t OFF_S = OFF_VB + (size_t)M_TOK * DM * 2;      // f32 chunk states
static constexpr size_t OFF_N = OFF_S + (size_t)NCHUNK * DM * DM * 4; // f32 numerator/logits
static constexpr size_t OFF_DEN = OFF_N + (size_t)M_TOK * DM * 4;     // f32 denom

extern "C" void kernel_launch(void* const* d_in, const int* in_sizes, int n_in,
                              void* d_out, int out_size, void* d_ws, size_t ws_size,
                              hipStream_t stream) {
  (void)in_sizes; (void)n_in; (void)out_size; (void)ws_size;
  const float* x = (const float*)d_in[0];
  const float* in_W = (const float*)d_in[1];
  const float* in_b = (const float*)d_in[2];
  const float* Wq = (const float*)d_in[3];
  const float* Wk = (const float*)d_in[4];
  const float* Wv = (const float*)d_in[5];
  const float* h1_W = (const float*)d_in[6];
  const float* h1_b = (const float*)d_in[7];
  const float* h2_W = (const float*)d_in[8];
  const float* h2_b = (const float*)d_in[9];
  const float* out_W = (const float*)d_in[10];
  const float* out_b = (const float*)d_in[11];

  char* ws = (char*)d_ws;
  bf16_t* Wi_bf = (bf16_t*)(ws + OFF_WIN);
  bf16_t* Wq_bf = (bf16_t*)(ws + OFF_WQ);
  bf16_t* Wk_bf = (bf16_t*)(ws + OFF_WK);
  bf16_t* Wv_bf = (bf16_t*)(ws + OFF_WV);
  bf16_t* W1_bf = (bf16_t*)(ws + OFF_WH1);
  bf16_t* W2_bf = (bf16_t*)(ws + OFF_WH2);
  bf16_t* Wo_bf = (bf16_t*)(ws + OFF_WO);
  float* h = (float*)(ws + OFF_H);
  bf16_t* hb = (bf16_t*)(ws + OFF_HB);   // also score buffer & u buffer (disjoint lifetimes)
  bf16_t* Qb = (bf16_t*)(ws + OFF_QB);   // also xb
  bf16_t* Kb = (bf16_t*)(ws + OFF_KB);   // also FFN mid
  bf16_t* Vb = (bf16_t*)(ws + OFF_VB);
  float* Sbuf = (float*)(ws + OFF_S);
  float* Nbuf = (float*)(ws + OFF_N);
  float* denom = (float*)(ws + OFF_DEN);

  const dim3 blk(128);
  const dim3 gFull(DM / BLK_N, M_TOK / BLK_M, 1);       // 8 x 512
  const dim3 gChunk(CROWS / BLK_N, CROWS / BLK_M, NCHUNK);
  auto cvtGrid = [](size_t n) { return dim3((unsigned)((n + 255) / 256)); };

  // weights -> bf16 (deterministic every call)
  cvt_f32_bf16<<<cvtGrid((size_t)DM * DIO), 256, 0, stream>>>(in_W, Wi_bf, (size_t)DM * DIO);
  cvt_f32_bf16<<<cvtGrid((size_t)NLAYER * DM * DM), 256, 0, stream>>>(Wq, Wq_bf, (size_t)NLAYER * DM * DM);
  cvt_f32_bf16<<<cvtGrid((size_t)NLAYER * DM * DM), 256, 0, stream>>>(Wk, Wk_bf, (size_t)NLAYER * DM * DM);
  cvt_f32_bf16<<<cvtGrid((size_t)NLAYER * DM * DM), 256, 0, stream>>>(Wv, Wv_bf, (size_t)NLAYER * DM * DM);
  cvt_f32_bf16<<<cvtGrid((size_t)NLAYER * DM * DM), 256, 0, stream>>>(h1_W, W1_bf, (size_t)NLAYER * DM * DM);
  cvt_f32_bf16<<<cvtGrid((size_t)NLAYER * DM * DM), 256, 0, stream>>>(h2_W, W2_bf, (size_t)NLAYER * DM * DM);
  cvt_f32_bf16<<<cvtGrid((size_t)DIO * DM), 256, 0, stream>>>(out_W, Wo_bf, (size_t)DIO * DM);
  cvt_f32_bf16<<<cvtGrid((size_t)M_TOK * DIO), 256, 0, stream>>>(x, Qb, (size_t)M_TOK * DIO); // xb

  // h = relu(x @ in_W.T + b) + pe  -> h (f32) + hb (bf16)
  gemm_nt_staged<EPI_IN_PROJ><<<gFull, blk, 0, stream>>>(
      Qb, DIO, Wi_bf, DIO, h, hb, DM, in_b, nullptr, DIO, 0, 0, 0);

  for (int l = 0; l < NLAYER; ++l) {
    const bf16_t* Wq_l = Wq_bf + (size_t)l * DM * DM;
    const bf16_t* Wk_l = Wk_bf + (size_t)l * DM * DM;
    const bf16_t* Wv_l = Wv_bf + (size_t)l * DM * DM;
    const bf16_t* W1_l = W1_bf + (size_t)l * DM * DM;
    const bf16_t* W2_l = W2_bf + (size_t)l * DM * DM;
    const float* b1_l = h1_b + (size_t)l * DM;
    const float* b2_l = h2_b + (size_t)l * DM;

    // Q = elu(h@Wq.T)+1, K = elu(h@Wk.T)+1, V = h@Wv.T   (bf16)
    gemm_nt_staged<EPI_ELU1_BF16><<<gFull, blk, 0, stream>>>(
        hb, DM, Wq_l, DM, nullptr, Qb, DM, nullptr, nullptr, DM, 0, 0, 0);
    gemm_nt_staged<EPI_ELU1_BF16><<<gFull, blk, 0, stream>>>(
        hb, DM, Wk_l, DM, nullptr, Kb, DM, nullptr, nullptr, DM, 0, 0, 0);
    gemm_nt_staged<EPI_BF16><<<gFull, blk, 0, stream>>>(
        hb, DM, Wv_l, DM, nullptr, Vb, DM, nullptr, nullptr, DM, 0, 0, 0);

    denom_kernel<<<BATCH_, 256, 0, stream>>>(Qb, Kb, denom);

    // per-chunk KV states: S[c] = K_c^T @ V_c   (f32)
    gemm_wmma<AL_CM, BL_NN, EPI_F32><<<gChunk, blk, 0, stream>>>(
        Kb, DM, Vb, DM, Sbuf, nullptr, DM, nullptr, nullptr, CROWS,
        (long long)CROWS * DM, 0, (long long)CROWS * DM, 0, (long long)DM * DM, 0);
    // inclusive prefix over chunks
    for (int c = 1; c < NCHUNK; ++c)
      addip_f32<<<cvtGrid((size_t)DM * DM), 256, 0, stream>>>(
          Sbuf + (size_t)c * DM * DM, Sbuf + (size_t)(c - 1) * DM * DM, (size_t)DM * DM);

    // intra-chunk: scores = mask(Q_c @ K_c^T) -> bf16 (reuses hb region)
    gemm_nt_staged<EPI_MASK_BF16><<<gChunk, blk, 0, stream>>>(
        Qb, DM, Kb, DM, nullptr, hb, CROWS, nullptr, nullptr, DM,
        (long long)CROWS * DM, (long long)CROWS * DM, (long long)CROWS * CROWS);
    // numer = scores @ V_c
    gemm_wmma<AL_RM, BL_NN, EPI_F32><<<gChunk, blk, 0, stream>>>(
        hb, CROWS, Vb, DM, Nbuf, nullptr, DM, nullptr, nullptr, CROWS,
        (long long)CROWS * CROWS, 0, (long long)CROWS * DM, 0, (long long)CROWS * DM, 0);
    // numer += Q_c @ S[c-1]   (chunks 1..31; S is f32, converted in-register)
    gemm_wmma<AL_RM, BL_NN_F32, EPI_F32_ACC><<<dim3(CROWS / BLK_N, CROWS / BLK_M, NCHUNK - 1), blk, 0, stream>>>(
        Qb, DM, Sbuf, DM, Nbuf, nullptr, DM, nullptr, nullptr, DM,
        (long long)CROWS * DM, 1, (long long)DM * DM, 0, (long long)CROWS * DM, 1);

    // u = numer/denom + h   (bf16, reuses hb region)
    combine_kernel<<<cvtGrid((size_t)M_TOK * DM), 256, 0, stream>>>(
        Nbuf, denom, h, hb, (size_t)M_TOK * DM);

    // FFN: mid = relu(u@W1.T+b1) (reuses Kb);  h = mid@W2.T + b2 + h -> h,hb
    gemm_nt_staged<EPI_BIAS_RELU_BF16><<<gFull, blk, 0, stream>>>(
        hb, DM, W1_l, DM, nullptr, Kb, DM, b1_l, nullptr, DM, 0, 0, 0);
    gemm_nt_staged<EPI_FFN2><<<gFull, blk, 0, stream>>>(
        Kb, DM, W2_l, DM, h, hb, DM, b2_l, h, DM, 0, 0, 0);
  }

  // logits = h @ out_W.T + out_b (f32 into Nbuf), then row softmax -> d_out
  gemm_nt_staged<EPI_BIAS_F32><<<dim3(DIO / BLK_N, M_TOK / BLK_M, 1), blk, 0, stream>>>(
      hb, DM, Wo_bf, DM, Nbuf, nullptr, DIO, out_b, nullptr, DM, 0, 0, 0);
  softmax_kernel<<<M_TOK, 256, 0, stream>>>(Nbuf, (float*)d_out);
}
